// MultiScaleLDSA_20100446945661
// MI455X (gfx1250) — matrix operand order repaired and checked by reading, not compiled, not run on hardware
//
#include <hip/hip_runtime.h>
#include <hip/hip_bf16.h>

// ---------------------------------------------------------------------------
// MultiScaleLDSA for MI455X (gfx1250): bf16 WMMA GEMMs + scalar local attention
// GEMM waves compute 32(M) x 64(N): 2 A-fragments x 4 B-fragments = 8 WMMA/K-step
// ---------------------------------------------------------------------------

typedef __attribute__((ext_vector_type(16))) __bf16 v16bf;
typedef __attribute__((ext_vector_type(8)))  __bf16 v8bf;
typedef __attribute__((ext_vector_type(8)))  float  v8f;

#define NF     512          // N_FEAT
#define BB     4            // batch
#define TT     2048         // sequence
#define MROWS  (BB*TT)      // 8192 tokens
#define HS     4            // heads per scale
#define HD     128          // head dim (NF / HS)
#define CS0    31
#define CS1    61
#define NP0    128          // padded logit width, scale 0 (HS*CS0 = 124)
#define NP1    256          // padded logit width, scale 1 (HS*CS1 = 244)

// ---------------------------------------------------------------------------
// Elementwise f32 -> bf16 conversion
// ---------------------------------------------------------------------------
__global__ void k_f32_to_bf16(const float* __restrict__ src,
                              __bf16* __restrict__ dst, int n) {
  int i = blockIdx.x * blockDim.x + threadIdx.x;
  if (i < n) dst[i] = (__bf16)src[i];
}

// f32 [rows,K] -> bf16 [rowsPad,K], zero-padding extra rows
__global__ void k_pad_bf16(const float* __restrict__ src,
                           __bf16* __restrict__ dst,
                           int rows, int rowsPad, int K) {
  int i = blockIdx.x * blockDim.x + threadIdx.x;
  if (i >= rowsPad * K) return;
  int r = i / K;
  dst[i] = (r < rows) ? (__bf16)src[i] : (__bf16)0.0f;
}

// ---------------------------------------------------------------------------
// bf16 WMMA GEMM:  C[M,N](f32 or bf16, optional ReLU) = A[M,K] * B[N,K]^T
// Block = 128 threads = 4 waves. Each wave computes a 32(M) x 64(N) strip,
// block covers 128(M) x 64(N).  Grid: x = N/64, y = M/128.
// Per k-step: 2 A fragments x 4 B fragments -> 8 v_wmma_f32_16x16x32_bf16.
// M%128==0, N%64==0, K%32==0 assumed (all call sites satisfy this).
// ---------------------------------------------------------------------------
template <bool RELU, bool OUT_BF16>
__global__ __launch_bounds__(128)
void k_gemm_bf16(const __bf16* __restrict__ A,
                 const __bf16* __restrict__ Bw,
                 void* __restrict__ Cout,
                 int M, int N, int K, int ldc) {
  const int lane   = threadIdx.x & 31;
  const int wave   = threadIdx.x >> 5;
  const int l15    = lane & 15;
  const int half   = lane >> 4;                 // 0: lanes 0-15, 1: lanes 16-31
  const int mBase  = blockIdx.y * 128 + wave * 32;
  const int nBase  = blockIdx.x * 64;

  // A fragment per ISA 7.12.2 (16-bit A 16x32): lanes 0-15 hold M=0..15 with
  // K {0..7,16..23}; lanes 16-31 hold same M with K {8..15,24..31}.
  const __bf16* aPtr0 = A + (size_t)(mBase + l15) * K + (half ? 8 : 0);
  const __bf16* aPtr1 = A + (size_t)(mBase + 16 + l15) * K + (half ? 8 : 0);
  // B fragment (16-bit B 32x16): lane = N (mod 16), same K order as A:
  // lanes 0-15 K {0..7,16..23}, lanes 16-31 K {8..15,24..31}.
  const __bf16* bPtr[4];
#pragma unroll
  for (int nt = 0; nt < 4; ++nt)
    bPtr[nt] = Bw + (size_t)(nBase + nt * 16 + l15) * K + (half ? 8 : 0);

  v8f acc[2][4];
#pragma unroll
  for (int mt = 0; mt < 2; ++mt)
#pragma unroll
    for (int nt = 0; nt < 4; ++nt)
      acc[mt][nt] = v8f{};

  for (int k0 = 0; k0 < K; k0 += 32) {
    __builtin_prefetch(aPtr0 + k0 + 64, 0, 1);  // next A chunks -> global_prefetch
    __builtin_prefetch(aPtr1 + k0 + 64, 0, 1);
    union { v16bf v; v8bf h[2]; } a0, a1;
    a0.h[0] = *(const v8bf*)(aPtr0 + k0);
    a0.h[1] = *(const v8bf*)(aPtr0 + k0 + 16);
    a1.h[0] = *(const v8bf*)(aPtr1 + k0);
    a1.h[1] = *(const v8bf*)(aPtr1 + k0 + 16);
#pragma unroll
    for (int nt = 0; nt < 4; ++nt) {
      union { v16bf v; v8bf h[2]; } bu;
      bu.h[0] = *(const v8bf*)(bPtr[nt] + k0);
      bu.h[1] = *(const v8bf*)(bPtr[nt] + k0 + 16);
      v16bf b = bu.v;
      acc[0][nt] = __builtin_amdgcn_wmma_f32_16x16x32_bf16(
          false, a0.v, false, b, (short)0, acc[0][nt], false, false);
      acc[1][nt] = __builtin_amdgcn_wmma_f32_16x16x32_bf16(
          false, a1.v, false, b, (short)0, acc[1][nt], false, false);
    }
  }

  // C/D layout: VGPR j -> M = half*8 + j, N = lane&15 within each 16x16 tile.
  float*  Cf = (float*)Cout;
  __bf16* Cb = (__bf16*)Cout;
#pragma unroll
  for (int mt = 0; mt < 2; ++mt) {
    const int rowOff = mBase + mt * 16 + (half ? 8 : 0);
#pragma unroll
    for (int nt = 0; nt < 4; ++nt) {
      const int col = nBase + nt * 16 + l15;
#pragma unroll
      for (int j = 0; j < 8; ++j) {
        float val = acc[mt][nt][j];
        if (RELU) val = val > 0.0f ? val : 0.0f;
        const size_t idx = (size_t)(rowOff + j) * ldc + col;
        if (OUT_BF16) Cb[idx] = (__bf16)val; else Cf[idx] = val;
      }
    }
  }
}

// ---------------------------------------------------------------------------
// Per-(token,head) softmax over cs logits. logits row stride = ldl (padded).
// attn written densely: [token, head, cs].
// ---------------------------------------------------------------------------
__global__ void k_softmax(const float* __restrict__ logits,
                          float* __restrict__ attn, int cs, int ldl) {
  int r = blockIdx.x * blockDim.x + threadIdx.x;   // r = token*HS + h
  if (r >= MROWS * HS) return;
  const int token = r / HS;
  const int h     = r - token * HS;
  const float* src = logits + (size_t)token * ldl + h * cs;
  float* dst = attn + (size_t)r * cs;
  float mx = -1e30f;
  for (int c = 0; c < cs; ++c) mx = fmaxf(mx, src[c]);
  float sum = 0.0f;
  for (int c = 0; c < cs; ++c) {
    float e = __expf(src[c] - mx);
    dst[c] = e;
    sum += e;
  }
  float inv = 1.0f / sum;
  for (int c = 0; c < cs; ++c) dst[c] *= inv;
}

// ---------------------------------------------------------------------------
// Windowed aggregation, both scales fused + scale-weight softmax.
// One 512-thread block per token; attention weights staged in LDS.
// out[token, f] = sw0 * sum_c a0[h,c]*v0[t+c-15, f] + sw1 * sum_c a1[h,c]*v1[t+c-30, f]
// ---------------------------------------------------------------------------
__global__ __launch_bounds__(512)
void k_aggregate(const float* __restrict__ attn0, const float* __restrict__ attn1,
                 const float* __restrict__ v0,    const float* __restrict__ v1,
                 const float* __restrict__ scaleW, __bf16* __restrict__ outc) {
  __shared__ float la0[HS * CS0];   // 124 floats
  __shared__ float la1[HS * CS1];   // 244 floats
  const int token = blockIdx.x;
  const int b = token / TT;
  const int t = token - b * TT;
  const int f = threadIdx.x;

  for (int i = f; i < HS * CS0; i += 512) la0[i] = attn0[(size_t)token * (HS * CS0) + i];
  for (int i = f; i < HS * CS1; i += 512) la1[i] = attn1[(size_t)token * (HS * CS1) + i];
  __syncthreads();

  const int h = f >> 7;
  float acc0 = 0.0f, acc1 = 0.0f;
  for (int c = 0; c < CS0; ++c) {
    int tt = t + c - (CS0 - 1) / 2;
    if (tt >= 0 && tt < TT)
      acc0 += la0[h * CS0 + c] * v0[((size_t)b * TT + tt) * NF + f];
  }
  for (int c = 0; c < CS1; ++c) {
    int tt = t + c - (CS1 - 1) / 2;
    if (tt >= 0 && tt < TT)
      acc1 += la1[h * CS1 + c] * v1[((size_t)b * TT + tt) * NF + f];
  }
  const float e0 = __expf(scaleW[0]);
  const float e1 = __expf(scaleW[1]);
  const float inv = 1.0f / (e0 + e1);
  outc[(size_t)token * NF + f] = (__bf16)((e0 * inv) * acc0 + (e1 * inv) * acc1);
}

// ---------------------------------------------------------------------------
// Host-side orchestration
// ---------------------------------------------------------------------------
extern "C" void kernel_launch(void* const* d_in, const int* in_sizes, int n_in,
                              void* d_out, int out_size, void* d_ws, size_t ws_size,
                              hipStream_t stream) {
  const float* q    = (const float*)d_in[0];
  // d_in[1] = key (unused by the reference forward)
  const float* v    = (const float*)d_in[2];
  const float* w1_0 = (const float*)d_in[3];
  const float* w1_1 = (const float*)d_in[4];
  const float* w2_0 = (const float*)d_in[5];
  const float* w2_1 = (const float*)d_in[6];
  const float* w3_0 = (const float*)d_in[7];
  const float* w3_1 = (const float*)d_in[8];
  const float* swgt = (const float*)d_in[9];
  const float* wout = (const float*)d_in[10];

  char* ws = (char*)d_ws;
  size_t off = 0;
  auto alloc = [&](size_t bytes) -> void* {
    off = (off + 255) & ~(size_t)255;
    void* p = ws + off;
    off += bytes;
    return p;
  };

  const size_t actE = (size_t)MROWS * NF;       // 4,194,304 elements
  __bf16* qA      = (__bf16*)alloc(actE * 2);
  __bf16* vA      = (__bf16*)alloc(actE * 2);
  __bf16* w1b[2]  = {(__bf16*)alloc(NF * NF * 2), (__bf16*)alloc(NF * NF * 2)};
  __bf16* w3b[2]  = {(__bf16*)alloc(NF * NF * 2), (__bf16*)alloc(NF * NF * 2)};
  __bf16* woutb   = (__bf16*)alloc(NF * NF * 2);
  __bf16* w2b0    = (__bf16*)alloc((size_t)NP0 * NF * 2);
  __bf16* w2b1    = (__bf16*)alloc((size_t)NP1 * NF * 2);
  __bf16* qrelu0  = (__bf16*)alloc(actE * 2);
  __bf16* qrelu1  = (__bf16*)alloc(actE * 2);
  float*  v0f     = (float*)alloc(actE * 4);
  float*  v1f     = (float*)alloc(actE * 4);
  float*  logits0 = (float*)alloc((size_t)MROWS * NP0 * 4);
  float*  logits1 = (float*)alloc((size_t)MROWS * NP1 * 4);
  float*  attn0   = (float*)alloc((size_t)MROWS * HS * CS0 * 4);
  float*  attn1   = (float*)alloc((size_t)MROWS * HS * CS1 * 4);
  __bf16* comb    = (__bf16*)alloc(actE * 2);

  const dim3 blk(256);
  // --- conversions to bf16 ---
  k_f32_to_bf16<<<dim3((actE + 255) / 256), blk, 0, stream>>>(q, qA, (int)actE);
  k_f32_to_bf16<<<dim3((actE + 255) / 256), blk, 0, stream>>>(v, vA, (int)actE);
  const int wE = NF * NF;
  k_f32_to_bf16<<<dim3((wE + 255) / 256), blk, 0, stream>>>(w1_0, w1b[0], wE);
  k_f32_to_bf16<<<dim3((wE + 255) / 256), blk, 0, stream>>>(w1_1, w1b[1], wE);
  k_f32_to_bf16<<<dim3((wE + 255) / 256), blk, 0, stream>>>(w3_0, w3b[0], wE);
  k_f32_to_bf16<<<dim3((wE + 255) / 256), blk, 0, stream>>>(w3_1, w3b[1], wE);
  k_f32_to_bf16<<<dim3((wE + 255) / 256), blk, 0, stream>>>(wout, woutb, wE);
  k_pad_bf16<<<dim3((NP0 * NF + 255) / 256), blk, 0, stream>>>(w2_0, w2b0, HS * CS0, NP0, NF);
  k_pad_bf16<<<dim3((NP1 * NF + 255) / 256), blk, 0, stream>>>(w2_1, w2b1, HS * CS1, NP1, NF);

  const dim3 gblk(128);                    // 4 waves, 32x64 tile each
  const dim3 g512(NF / 64, MROWS / 128);   // (8, 64)
  const dim3 g128(NP0 / 64, MROWS / 128);  // (2, 64)
  const dim3 g256(NP1 / 64, MROWS / 128);  // (4, 64)

  // --- scale 0 ---
  k_gemm_bf16<true,  true ><<<g512, gblk, 0, stream>>>(qA, w1b[0], qrelu0, MROWS, NF, NF, NF);
  k_gemm_bf16<false, false><<<g512, gblk, 0, stream>>>(vA, w3b[0], v0f,    MROWS, NF, NF, NF);
  k_gemm_bf16<false, false><<<g128, gblk, 0, stream>>>(qrelu0, w2b0, logits0, MROWS, NP0, NF, NP0);
  k_softmax<<<dim3((MROWS * HS + 255) / 256), blk, 0, stream>>>(logits0, attn0, CS0, NP0);

  // --- scale 1 ---
  k_gemm_bf16<true,  true ><<<g512, gblk, 0, stream>>>(qA, w1b[1], qrelu1, MROWS, NF, NF, NF);
  k_gemm_bf16<false, false><<<g512, gblk, 0, stream>>>(vA, w3b[1], v1f,    MROWS, NF, NF, NF);
  k_gemm_bf16<false, false><<<g256, gblk, 0, stream>>>(qrelu1, w2b1, logits1, MROWS, NP1, NF, NP1);
  k_softmax<<<dim3((MROWS * HS + 255) / 256), blk, 0, stream>>>(logits1, attn1, CS1, NP1);

  // --- windowed aggregation + scale mix ---
  k_aggregate<<<dim3(MROWS), dim3(512), 0, stream>>>(attn0, attn1, v0f, v1f, swgt, comb);

  // --- output projection, f32 straight into d_out ---
  k_gemm_bf16<false, false><<<g512, gblk, 0, stream>>>(comb, woutb, d_out, MROWS, NF, NF, NF);

  (void)in_sizes; (void)n_in; (void)out_size; (void)ws_size;
}
